// MPNN_223338299453
// MI455X (gfx1250) — matrix operand order repaired and checked by reading, not compiled, not run on hardware
//
#include <hip/hip_runtime.h>
#include <hip/hip_bf16.h>

// ---------------------------------------------------------------------------
// Types for WMMA fragments (CDNA5 / gfx1250, wave32)
// ---------------------------------------------------------------------------
typedef __attribute__((ext_vector_type(16))) __bf16       v16bf;
typedef __attribute__((ext_vector_type(8)))  float        v8f;
typedef __attribute__((ext_vector_type(4)))  unsigned int v4u;
typedef __attribute__((ext_vector_type(4)))  unsigned int u32x4;
typedef __attribute__((ext_vector_type(8)))  int          i32x8;
typedef __attribute__((ext_vector_type(4)))  int          i32x4;

union BF16Frag {
  v16bf          v;
  v4u            q[2];
  unsigned int   u[8];
  unsigned short s[16];
};

__device__ __forceinline__ unsigned short f2bf(float f) {
  unsigned u = __builtin_bit_cast(unsigned, f);
  u += 0x7FFFu + ((u >> 16) & 1u);   // round-to-nearest-even
  return (unsigned short)(u >> 16);
}

// Packed f32x2 -> bf16x2: single v_cvt_pk_bf16_f32 (builtin not declared on
// this toolchain; the instruction exists on gfx1250 -> inline asm).
// D[15:0] = bf16(a), D[31:16] = bf16(b), round-to-nearest-even.
__device__ __forceinline__ unsigned cvt2(float a, float b) {
  unsigned r;
  asm("v_cvt_pk_bf16_f32 %0, %1, %2" : "=v"(r) : "v"(a), "v"(b));
  return r;
}

__device__ __forceinline__ void cvt8(const float* __restrict__ p,
                                     unsigned int* d) {
  float4 a = *(const float4*)p;
  float4 b = *(const float4*)(p + 4);
  d[0] = cvt2(a.x, a.y);
  d[1] = cvt2(a.z, a.w);
  d[2] = cvt2(b.x, b.y);
  d[3] = cvt2(b.z, b.w);
}

// ---------------------------------------------------------------------------
// LDS B-tile staging via the Tensor Data Mover.
// Layout: chunk c (32 K-values) x row n (64 N-rows), row pitch 80 B
// (64 B data + 16 B TDM pad -> 20-DWORD pitch => conflict-free 16-lane reads).
// ---------------------------------------------------------------------------
#define BROW_US 40  // 80 bytes / 2

template <int K>
__device__ __forceinline__ void stage_B_tile(unsigned short* Bs,
                                             const unsigned short* __restrict__ Bt,
                                             int nTile) {
#if __has_builtin(__builtin_amdgcn_tensor_load_to_lds) && \
    __has_builtin(__builtin_amdgcn_s_wait_tensorcnt)
  if (threadIdx.x < 32) {  // wave 0 issues the TDM DMA (wave-uniform branch)
    const unsigned long long ga =
        (unsigned long long)(const void*)(Bt + (size_t)nTile * K);
    const unsigned ldsAddr = (unsigned)(size_t)(void*)Bs;  // low 32b = LDS off

    u32x4 g0;
    g0.x = 1u;                                   // count=1, user descriptor
    g0.y = ldsAddr;                              // lds_addr
    g0.z = (unsigned)(ga & 0xFFFFFFFFu);         // global_addr[31:0]
    g0.w = (unsigned)((ga >> 32) & 0x01FFFFFFu)  // global_addr[56:32]
           | (2u << 30);                         // type = 2 ("image")

    i32x8 g1;
    g1[0] = (int)((1u << 16)      // data_size = 2 bytes
                  | (1u << 20)    // pad_enable
                  | (3u << 22)    // pad_interval: 16 DWORDs (one 64B row)
                  | (3u << 25));  // pad_amount:   4 DWORDs (16 B)
    g1[1] = (int)(32u << 16);     // tensor_dim0 = 32 (low16 in [31:16])
    g1[2] = (int)(64u << 16);     // tensor_dim0 hi=0 | tensor_dim1 = 64
    g1[3] = (int)(32u << 16);     // tensor_dim1 hi=0 | tile_dim0 = 32
    g1[4] = (int)(64u | ((unsigned)(K / 32) << 16));  // tile_dim1|tile_dim2
    g1[5] = (int)K;               // tensor_dim0_stride = K (between rows)
    g1[6] = (int)(32u << 16);     // stride0 hi=0 | tensor_dim1_stride = 32
    g1[7] = 0;                    // stride1 hi = 0

    i32x4 g2;
    g2.x = K / 32; g2.y = 0; g2.z = 0; g2.w = 0;  // tensor_dim2 = #chunks
    i32x4 g3;
    g3.x = 0; g3.y = 0; g3.z = 0; g3.w = 0;
    i32x8 g4;                     // trailing 256-bit operand (6-arg form)
    g4[0] = 0; g4[1] = 0; g4[2] = 0; g4[3] = 0;
    g4[4] = 0; g4[5] = 0; g4[6] = 0; g4[7] = 0;

    __builtin_amdgcn_tensor_load_to_lds(g0, g1, g2, g3, g4, 0);
    __builtin_amdgcn_s_wait_tensorcnt(0);
  }
  __syncthreads();
#else
  // Fallback: cooperative copy into the same padded layout (16 B per step).
  for (int idx = threadIdx.x; idx < (K / 32) * 64 * 4; idx += 128) {
    const int c = idx >> 8, rem = idx & 255, n = rem >> 2, q = rem & 3;
    *(v4u*)(Bs + (size_t)(c * 64 + n) * BROW_US + q * 8) =
        *(const v4u*)(Bt + (size_t)(nTile + n) * K + c * 32 + q * 8);
  }
  __syncthreads();
#endif
}

// ---------------------------------------------------------------------------
// Fused gather + GEMM + bias(+ReLU) via v_wmma_f32_16x16x32_bf16.
// Block: 128 threads (4 waves). Block tile 64(M) x 64(N); wave tile 16 x 64.
// B tile staged in LDS by the TDM; fragments read with ds_load_b128.
// ---------------------------------------------------------------------------
enum GemmMode { MODE_EDGE = 0, MODE_NODE = 1, MODE_GLOB = 2, MODE_PLAIN = 3 };

template <int K>
__device__ __forceinline__ void wmma_row(v8f* acc, const BF16Frag& af, int c,
                                         const unsigned short* Bs, int mrow,
                                         int khalf) {
#pragma unroll
  for (int j = 0; j < 4; ++j) {
    const unsigned short* bp =
        Bs + (size_t)(c * 64 + j * 16 + mrow) * BROW_US + khalf * 8;
    BF16Frag bfr;
    bfr.q[0] = *(const v4u*)(bp);
    bfr.q[1] = *(const v4u*)(bp + 16);
    acc[j] = __builtin_amdgcn_wmma_f32_16x16x32_bf16(
        false, af.v, false, bfr.v, (short)0, acc[j], false, false);
  }
}

template <int MODE, bool RELU_BF16, int K>
__global__ void __launch_bounds__(128)
gemm_wmma(const float* __restrict__ g0, const float* __restrict__ g1,
          const float* __restrict__ g2,
          const int* __restrict__ row, const int* __restrict__ col,
          const int* __restrict__ batch,
          const unsigned short* __restrict__ Abf,
          const unsigned short* __restrict__ Bt,
          const float* __restrict__ bias,
          unsigned short* __restrict__ outBf, float* __restrict__ outF,
          int M, int N) {
  extern __shared__ unsigned short Bs[];  // (K/32) * 64 * 80 bytes

  const int lane  = threadIdx.x & 31;
  const int wave  = threadIdx.x >> 5;
  const int mrow  = lane & 15;
  const int khalf = lane >> 4;
  const int mBase = blockIdx.x * 64 + wave * 16;
  const int nTile = blockIdx.y * 64;

  const int m  = mBase + mrow;
  const int mc = (m < M) ? m : (M - 1);

  // Stage the 64 x K weight tile into LDS once (reused by whole K loop).
  stage_B_tile<K>(Bs, Bt, nTile);

  // Per-lane gather base pointers (128-float segments of the concat input).
  const float *p0 = nullptr, *p1 = nullptr, *p2 = nullptr, *p3 = nullptr;
  if constexpr (MODE == MODE_EDGE) {
    const int r = row[mc], c = col[mc];
    p0 = g0 + (size_t)r * 128;          // nodes[row]
    p1 = g0 + (size_t)c * 128;          // nodes[col]
    p2 = g1 + (size_t)mc * 128;         // edge_attr
    p3 = g2 + (size_t)batch[r] * 128;   // u[batch[row]]
  } else if constexpr (MODE == MODE_NODE) {
    p0 = g0 + (size_t)mc * 128;         // nodes
    p1 = g1 + (size_t)mc * 128;         // agg (scatter-mean of edges)
    p2 = g2 + (size_t)batch[mc] * 128;  // u[batch]
  } else if constexpr (MODE == MODE_GLOB) {
    p0 = g0 + (size_t)mc * 128;         // u
    p1 = g1 + (size_t)mc * 128;         // node mean
    p2 = g2 + (size_t)mc * 128;         // edge mean
  }

  v8f acc[4] = {};

  if constexpr (MODE == MODE_PLAIN) {
#pragma unroll
    for (int c = 0; c < K / 32; ++c) {
      BF16Frag af;
      const v4u* ap = (const v4u*)(Abf + (size_t)mc * K + c * 32 + khalf * 8);
      af.q[0] = ap[0];   // K = c*32+khalf*8 .. +7
      af.q[1] = ap[2];   // K = c*32+khalf*8+16 .. +23
      wmma_row<K>(acc, af, c, Bs, mrow, khalf);
    }
  } else {
    // Region-split K loop: base pointer resolved at compile time (no branchy
    // per-iteration select). Concat regions are 128-wide.
    const float* bases[4] = {p0, p1, p2, (MODE == MODE_EDGE) ? p3 : p2};
#pragma unroll
    for (int r = 0; r < K / 128; ++r) {
      const float* base = bases[r];
#pragma unroll
      for (int s = 0; s < 4; ++s) {
        const int c = r * 4 + s;
        BF16Frag af;
        cvt8(base + s * 32 + khalf * 8, af.u);
        cvt8(base + s * 32 + khalf * 8 + 16, af.u + 4);
        wmma_row<K>(acc, af, c, Bs, mrow, khalf);
      }
    }
  }

  // Epilogue: C/D layout -> lane n = lane&15, rows = khalf*8 + v.
#pragma unroll
  for (int j = 0; j < 4; ++j) {
    const int n = nTile + j * 16 + mrow;
    const float bv = bias[n];
#pragma unroll
    for (int v = 0; v < 8; ++v) {
      const int mm = mBase + khalf * 8 + v;
      if (mm < M) {
        float val = acc[j][v] + bv;
        if constexpr (RELU_BF16) {
          val = (val > 0.f) ? val : 0.f;
          outBf[(size_t)mm * N + n] = f2bf(val);
        } else {
          outF[(size_t)mm * N + n] = val;
        }
      }
    }
  }
}

// ---------------------------------------------------------------------------
// Support kernels
// ---------------------------------------------------------------------------
__global__ void k_zero(float* p, int n) {
  int i = blockIdx.x * 256 + threadIdx.x;
  if (i < n) p[i] = 0.f;
}

// W [K,N] fp32 row-major  ->  Wt [N,K] bf16 (WMMA B operand friendly)
__global__ void k_convert_wt(const float* __restrict__ W,
                             unsigned short* __restrict__ Wt, int K, int N) {
  int idx = blockIdx.x * 256 + threadIdx.x;
  if (idx >= K * N) return;
  int n = idx / K, k = idx % K;
  Wt[idx] = f2bf(W[(size_t)k * N + n]);
}

__global__ void k_counts_edges(const int* __restrict__ row,
                               const int* __restrict__ col,
                               const int* __restrict__ batch,
                               float* ncnt, float* gecnt, int E) {
  int e = blockIdx.x * 256 + threadIdx.x;
  if (e >= E) return;
  atomicAdd(&ncnt[col[e]], 1.f);
  atomicAdd(&gecnt[batch[row[e]]], 1.f);
}

__global__ void k_counts_nodes(const int* __restrict__ batch, float* gncnt,
                               int NN) {
  int i = blockIdx.x * 256 + threadIdx.x;
  if (i >= NN) return;
  atomicAdd(&gncnt[batch[i]], 1.f);
}

__global__ void k_edge_scatter(const float* __restrict__ eo,
                               const int* __restrict__ row,
                               const int* __restrict__ col,
                               const int* __restrict__ batch,
                               float* agg, float* gesum, int E) {
  int idx = blockIdx.x * 256 + threadIdx.x;
  if (idx >= E * 128) return;
  int e = idx >> 7, k = idx & 127;
  float v = eo[idx];
  atomicAdd(&agg[(size_t)col[e] * 128 + k], v);
  atomicAdd(&gesum[(size_t)batch[row[e]] * 128 + k], v);
}

__global__ void k_agg_div(float* agg, const float* __restrict__ ncnt, int NN) {
  int idx = blockIdx.x * 256 + threadIdx.x;
  if (idx >= NN * 128) return;
  float c = ncnt[idx >> 7];
  agg[idx] /= (c > 1.f ? c : 1.f);
}

__global__ void k_node_scatter(const float* __restrict__ no,
                               const int* __restrict__ batch, float* gnsum,
                               int NN) {
  int idx = blockIdx.x * 256 + threadIdx.x;
  if (idx >= NN * 128) return;
  int i = idx >> 7, k = idx & 127;
  atomicAdd(&gnsum[(size_t)batch[i] * 128 + k], no[idx]);
}

__global__ void k_gmean_div(float* gnsum, const float* gncnt, float* gesum,
                            const float* gecnt, int tot) {
  int idx = blockIdx.x * 256 + threadIdx.x;
  if (idx >= tot) return;
  int g = idx >> 7;
  float cn = gncnt[g], ce = gecnt[g];
  gnsum[idx] /= (cn > 1.f ? cn : 1.f);
  gesum[idx] /= (ce > 1.f ? ce : 1.f);
}

// ---------------------------------------------------------------------------
// Host launcher
// ---------------------------------------------------------------------------
static inline size_t rnd256(size_t x) { return (x + 255) & ~(size_t)255; }
static inline size_t ldsBytes(int K) { return (size_t)(K / 32) * 64 * 80; }

extern "C" void kernel_launch(void* const* d_in, const int* in_sizes, int n_in,
                              void* d_out, int out_size, void* d_ws,
                              size_t ws_size, hipStream_t stream) {
  const float* nodes = (const float*)d_in[0];
  const int*   eidx  = (const int*)d_in[1];
  const float* eattr = (const float*)d_in[2];
  const float* u     = (const float*)d_in[3];
  const int*   batch = (const int*)d_in[4];

  const float* ew[3] = {(const float*)d_in[5], (const float*)d_in[7],
                        (const float*)d_in[9]};
  const float* eb[3] = {(const float*)d_in[6], (const float*)d_in[8],
                        (const float*)d_in[10]};
  const float* nw[3] = {(const float*)d_in[11], (const float*)d_in[13],
                        (const float*)d_in[15]};
  const float* nb[3] = {(const float*)d_in[12], (const float*)d_in[14],
                        (const float*)d_in[16]};
  const float* gw[3] = {(const float*)d_in[17], (const float*)d_in[19],
                        (const float*)d_in[21]};
  const float* gb[3] = {(const float*)d_in[18], (const float*)d_in[20],
                        (const float*)d_in[22]};

  const int D = 128, H = 256;
  const int E  = in_sizes[1] / 2;
  const int NN = in_sizes[0] / D;
  const int NG = in_sizes[3] / D;
  const int* row = eidx;
  const int* col = eidx + E;

  // -------- workspace layout --------
  char* ws = (char*)d_ws;
  size_t off = 0;
  auto alloc = [&](size_t bytes) {
    size_t o = off;
    off = rnd256(off + bytes);
    return o;
  };

  const int Ke[3] = {4 * D, H, H};
  const int Kn[3] = {3 * D, H, H};
  const int Nd[3] = {H, H, D};
  size_t wtE[3], wtN[3], wtG[3];
  for (int i = 0; i < 3; i++) wtE[i] = alloc((size_t)Ke[i] * Nd[i] * 2);
  for (int i = 0; i < 3; i++) wtN[i] = alloc((size_t)Kn[i] * Nd[i] * 2);
  for (int i = 0; i < 3; i++) wtG[i] = alloc((size_t)Kn[i] * Nd[i] * 2);

  size_t h0O   = alloc((size_t)E * H * 2);   // bf16 hidden activations
  size_t h1O   = alloc((size_t)E * H * 2);
  size_t aggO  = alloc((size_t)NN * D * 4);  // segment sums / means (f32)
  size_t ncntO = alloc((size_t)NN * 4);
  size_t geO   = alloc((size_t)NG * D * 4);
  size_t gecO  = alloc((size_t)NG * 4);
  size_t gnO   = alloc((size_t)NG * D * 4);
  size_t gncO  = alloc((size_t)NG * 4);
  size_t zbeg = aggO, zend = off;

  unsigned short* H0 = (unsigned short*)(ws + h0O);
  unsigned short* H1 = (unsigned short*)(ws + h1O);
  float* AGG   = (float*)(ws + aggO);
  float* NCNT  = (float*)(ws + ncntO);
  float* GESUM = (float*)(ws + geO);
  float* GECNT = (float*)(ws + gecO);
  float* GNSUM = (float*)(ws + gnO);
  float* GNCNT = (float*)(ws + gncO);

  float* out_nodes = (float*)d_out;
  float* out_edges = out_nodes + (size_t)NN * D;
  float* out_glob  = out_edges + (size_t)E * D;

  // -------- zero atomic accumulators (every call; ws is not re-poisoned) ---
  {
    int nf = (int)((zend - zbeg) / 4);
    k_zero<<<(nf + 255) / 256, 256, 0, stream>>>((float*)(ws + zbeg), nf);
  }

  // -------- convert + transpose weights to bf16 [N][K] --------
  auto conv = [&](const float* W, size_t o, int K, int N) {
    int tot = K * N;
    k_convert_wt<<<(tot + 255) / 256, 256, 0, stream>>>(
        W, (unsigned short*)(ws + o), K, N);
  };
  for (int i = 0; i < 3; i++) conv(ew[i], wtE[i], Ke[i], Nd[i]);
  for (int i = 0; i < 3; i++) conv(nw[i], wtN[i], Kn[i], Nd[i]);
  for (int i = 0; i < 3; i++) conv(gw[i], wtG[i], Kn[i], Nd[i]);

  const dim3 blk(128);
  auto grd = [](int M, int N) {
    return dim3((unsigned)((M + 63) / 64), (unsigned)(N / 64));
  };

  // -------- Edge MLP: gathered [E,512] -> 256 -> 256 -> 128 --------
  gemm_wmma<MODE_EDGE, true, 512><<<grd(E, H), blk, ldsBytes(512), stream>>>(
      nodes, eattr, u, row, col, batch, nullptr,
      (unsigned short*)(ws + wtE[0]), eb[0], H0, nullptr, E, H);
  gemm_wmma<MODE_PLAIN, true, 256><<<grd(E, H), blk, ldsBytes(256), stream>>>(
      nullptr, nullptr, nullptr, nullptr, nullptr, nullptr, H0,
      (unsigned short*)(ws + wtE[1]), eb[1], H1, nullptr, E, H);
  gemm_wmma<MODE_PLAIN, false, 256><<<grd(E, D), blk, ldsBytes(256), stream>>>(
      nullptr, nullptr, nullptr, nullptr, nullptr, nullptr, H1,
      (unsigned short*)(ws + wtE[2]), eb[2], nullptr, out_edges, E, D);

  // -------- segment means of edges_out --------
  k_counts_edges<<<(E + 255) / 256, 256, 0, stream>>>(row, col, batch, NCNT,
                                                      GECNT, E);
  k_counts_nodes<<<(NN + 255) / 256, 256, 0, stream>>>(batch, GNCNT, NN);
  k_edge_scatter<<<(E * D + 255) / 256, 256, 0, stream>>>(out_edges, row, col,
                                                          batch, AGG, GESUM, E);
  k_agg_div<<<(NN * D + 255) / 256, 256, 0, stream>>>(AGG, NCNT, NN);

  // -------- Node MLP: gathered [NN,384] -> 256 -> 256 -> 128 --------
  gemm_wmma<MODE_NODE, true, 384><<<grd(NN, H), blk, ldsBytes(384), stream>>>(
      nodes, AGG, u, nullptr, nullptr, batch, nullptr,
      (unsigned short*)(ws + wtN[0]), nb[0], H0, nullptr, NN, H);
  gemm_wmma<MODE_PLAIN, true, 256><<<grd(NN, H), blk, ldsBytes(256), stream>>>(
      nullptr, nullptr, nullptr, nullptr, nullptr, nullptr, H0,
      (unsigned short*)(ws + wtN[1]), nb[1], H1, nullptr, NN, H);
  gemm_wmma<MODE_PLAIN, false, 256><<<grd(NN, D), blk, ldsBytes(256), stream>>>(
      nullptr, nullptr, nullptr, nullptr, nullptr, nullptr, H1,
      (unsigned short*)(ws + wtN[2]), nb[2], nullptr, out_nodes, NN, D);

  // -------- per-graph means --------
  k_node_scatter<<<(NN * D + 255) / 256, 256, 0, stream>>>(out_nodes, batch,
                                                           GNSUM, NN);
  k_gmean_div<<<(NG * D + 255) / 256, 256, 0, stream>>>(GNSUM, GNCNT, GESUM,
                                                        GECNT, NG * D);

  // -------- Global MLP: gathered [NG,384] -> 256 -> 256 -> 128 --------
  gemm_wmma<MODE_GLOB, true, 384><<<grd(NG, H), blk, ldsBytes(384), stream>>>(
      u, GNSUM, GESUM, nullptr, nullptr, nullptr, nullptr,
      (unsigned short*)(ws + wtG[0]), gb[0], H0, nullptr, NG, H);
  gemm_wmma<MODE_PLAIN, true, 256><<<grd(NG, H), blk, ldsBytes(256), stream>>>(
      nullptr, nullptr, nullptr, nullptr, nullptr, nullptr, H0,
      (unsigned short*)(ws + wtG[1]), gb[1], H1, nullptr, NG, H);
  gemm_wmma<MODE_PLAIN, false, 256><<<grd(NG, D), blk, ldsBytes(256), stream>>>(
      nullptr, nullptr, nullptr, nullptr, nullptr, nullptr, H1,
      (unsigned short*)(ws + wtG[2]), gb[2], nullptr, out_glob, NG, D);

  (void)n_in; (void)out_size; (void)ws_size;
}
